// WindowAttention_54168127537767
// MI455X (gfx1250) — matrix operand order repaired and checked by reading, not compiled, not run on hardware
//
#include <hip/hip_runtime.h>
#include <hip/hip_bf16.h>
#include <stdint.h>

typedef __attribute__((ext_vector_type(16))) _Float16 v16h;
typedef __attribute__((ext_vector_type(8)))  float    v8f;
typedef __attribute__((ext_vector_type(4)))  _Float16 h4;
typedef __attribute__((ext_vector_type(4)))  uint32_t u32x4;

struct FragH { union { v16h v; u32x4 q[2]; uint32_t u[8]; }; };

// A-matrix 16x32 f16 fragment (ISA 7.12.2): lane m = lane&15, half h = lane>>4.
// VGPR j holds the f16 pair at k = k0 + 16*(j>>2) + 8*h + 2*(j&3).
// Two contiguous 16-byte-aligned chunks -> two 128-bit loads.
__device__ __forceinline__ v16h load_a16x32(const _Float16* __restrict__ src,
                                            int ld, int row0, int k0, int lane) {
    const int m = lane & 15, h = (lane >> 4) & 1;
    const _Float16* base = src + (row0 + m) * ld + k0 + 8 * h;
    FragH f;
    f.q[0] = *(const u32x4*)(base);
    f.q[1] = *(const u32x4*)(base + 16);
    return f.v;
}

// B-matrix 32x16 f16 fragment where B[k][n] = src[n][k] (src row-major).
// Lane n = lane&15, half h = lane>>4; VGPR j holds the pair at
// k = k0 + 16*(j>>2) + 8*h + 2*(j&3) -> two 128-bit loads.
__device__ __forceinline__ v16h load_bT32x16(const _Float16* __restrict__ src,
                                             int ld, int col0, int k0, int lane) {
    const int n = lane & 15, h = (lane >> 4) & 1;
    const _Float16* base = src + (col0 + n) * ld + k0 + 8 * h;
    FragH f;
    f.q[0] = *(const u32x4*)(base);
    f.q[1] = *(const u32x4*)(base + 16);
    return f.v;
}

__device__ __forceinline__ v8f wmma_f16(v16h a, v16h b, v8f c) {
    return __builtin_amdgcn_wmma_f32_16x16x32_f16(false, a, false, b,
                                                  (short)0, c, false, false);
}

// ---------------- weight f32 -> f16 pre-pass ----------------
__global__ void convert_weights_kernel(const float* __restrict__ qkv_w,
                                       const float* __restrict__ proj_w,
                                       _Float16* __restrict__ qkvw_h,
                                       _Float16* __restrict__ projw_h) {
    int i = blockIdx.x * blockDim.x + threadIdx.x;
    int stride = gridDim.x * blockDim.x;
    for (int j = i; j < 768 * 256; j += stride) qkvw_h[j] = (_Float16)qkv_w[j];
    for (int j = i; j < 256 * 256; j += stride) projw_h[j] = (_Float16)proj_w[j];
}

// ---------------- fused window attention, one block per window ----------------
// LDS layout (128 KB dynamic, 2 blocks/WGP fit in 320 KB):
//   [0,      32768) : Xh  f16 [64][256]  -- phase A/B; reused as S f32 [64][64]
//                     (first 16 KB) + P f16 [64][64] (at +16384) in phases C-E
//   [32768,  65536) : Qh  f16 [64][256]  -- reused as O1h in phases E/F
//   [65536,  98304) : Kh  f16 [64][256]
//   [98304, 131072) : VT  f16 [256][64]  (V transposed for contiguous B frags)
__global__ __launch_bounds__(256)
void WindowAttention_54168127537767_kernel(const float* __restrict__ x,
                                           const _Float16* __restrict__ qkvw_h,
                                           const float* __restrict__ qkv_b,
                                           const _Float16* __restrict__ projw_h,
                                           const float* __restrict__ proj_b,
                                           float* __restrict__ out) {
    extern __shared__ char smem[];
    _Float16* Xh = (_Float16*)(smem);
    float*    Sm = (float*)(smem);
    _Float16* Pm = (_Float16*)(smem + 16384);
    _Float16* Qh = (_Float16*)(smem + 32768);
    _Float16* Kh = (_Float16*)(smem + 65536);
    _Float16* VT = (_Float16*)(smem + 98304);
    _Float16* O1h = Qh;

    const int tid  = threadIdx.x;
    const int lane = tid & 31;
    const int wid  = tid >> 5;
    const int h    = (lane >> 4) & 1;
    const int n    = lane & 15;
    const int blk  = blockIdx.x;

    // ---- Phase A: load x[blk] f32 -> f16 into LDS (float4 / h4 vectorized)
    {
        const float4* xg = (const float4*)(x + (size_t)blk * 64 * 256);
        for (int i = tid; i < 64 * 256 / 4; i += 256) {
            float4 v = xg[i];
            h4 hv = { (_Float16)v.x, (_Float16)v.y, (_Float16)v.z, (_Float16)v.w };
            *(h4*)(Xh + 4 * i) = hv;
        }
    }
    __syncthreads();

    // ---- Phase B: QKV = Xh @ qkv_w^T + b (64x768). Walk 48 N-strips of 16
    // columns; 4 M-accumulators share each B fragment (1 B load -> 4 WMMAs).
    for (int s = wid; s < 48; s += 8) {
        v8f acc[4] = {};
#pragma unroll
        for (int kb = 0; kb < 8; ++kb) {
            v16h b = load_bT32x16(qkvw_h, 256, s * 16, kb * 32, lane);
#pragma unroll
            for (int tm = 0; tm < 4; ++tm) {
                v16h a = load_a16x32(Xh, 256, tm * 16, kb * 32, lane);
                acc[tm] = wmma_f16(a, b, acc[tm]);
            }
        }
        const int sec  = s >> 4;                  // 0:Q 1:K 2:V
        const int ecol = (s & 15) * 16 + n;       // column inside section
        const float bias = qkv_b[s * 16 + n];
#pragma unroll
        for (int tm = 0; tm < 4; ++tm) {
#pragma unroll
            for (int r = 0; r < 8; ++r) {
                const int row = tm * 16 + r + 8 * h;
                // branchless destination: Q/K row-major, V transposed
                const uint32_t off_qk = 32768u + (uint32_t)sec * 32768u
                                      + 2u * (uint32_t)(row * 256 + ecol);
                const uint32_t off_v  = 98304u + 2u * (uint32_t)(ecol * 64 + row);
                const uint32_t off = (sec == 2) ? off_v : off_qk;
                *(_Float16*)(smem + off) = (_Float16)(acc[tm][r] + bias);
            }
        }
    }
    __syncthreads();

    // ---- Phase C: S = Q @ K^T  (64x64 f32, 16 tiles, 2 per wave)
    for (int t = wid; t < 16; t += 8) {
        const int tm = t >> 2, tn = t & 3;
        v8f acc = {};
#pragma unroll
        for (int kb = 0; kb < 8; ++kb) {
            v16h a = load_a16x32(Qh, 256, tm * 16, kb * 32, lane);
            v16h b = load_bT32x16(Kh, 256, tn * 16, kb * 32, lane);
            acc = wmma_f16(a, b, acc);
        }
#pragma unroll
        for (int r = 0; r < 8; ++r)
            Sm[(tm * 16 + r + 8 * h) * 64 + tn * 16 + n] = acc[r];
    }
    __syncthreads();

    // ---- Phase D: row softmax (one thread per row), result f16 in Pm
    if (tid < 64) {
        const float* srow = Sm + tid * 64;
        float m = srow[0];
#pragma unroll 4
        for (int j = 1; j < 64; ++j) m = fmaxf(m, srow[j]);
        float s = 0.f;
        float e[64];
#pragma unroll 4
        for (int j = 0; j < 64; ++j) { e[j] = __expf(srow[j] - m); s += e[j]; }
        const float inv = 1.f / s;
#pragma unroll 4
        for (int j = 0; j < 64; ++j) Pm[tid * 64 + j] = (_Float16)(e[j] * inv);
    }
    __syncthreads();

    // ---- Phase E: O1 = P @ V (64x256, K=64): 16 N-strips, 4 M-accs each
    for (int s = wid; s < 16; s += 8) {
        v8f acc[4] = {};
#pragma unroll
        for (int kb = 0; kb < 2; ++kb) {
            v16h b = load_bT32x16(VT, 64, s * 16, kb * 32, lane); // B[k][n]=V[k][n]
#pragma unroll
            for (int tm = 0; tm < 4; ++tm) {
                v16h a = load_a16x32(Pm, 64, tm * 16, kb * 32, lane);
                acc[tm] = wmma_f16(a, b, acc[tm]);
            }
        }
#pragma unroll
        for (int tm = 0; tm < 4; ++tm)
#pragma unroll
            for (int r = 0; r < 8; ++r)
                O1h[(tm * 16 + r + 8 * h) * 256 + s * 16 + n] = (_Float16)acc[tm][r];
    }
    __syncthreads();

    // ---- Phase F: out = O1 @ proj_w^T + proj_b: 16 N-strips, 4 M-accs each
    float* outg = out + (size_t)blk * 64 * 256;
    for (int s = wid; s < 16; s += 8) {
        v8f acc[4] = {};
#pragma unroll
        for (int kb = 0; kb < 8; ++kb) {
            v16h b = load_bT32x16(projw_h, 256, s * 16, kb * 32, lane);
#pragma unroll
            for (int tm = 0; tm < 4; ++tm) {
                v16h a = load_a16x32(O1h, 256, tm * 16, kb * 32, lane);
                acc[tm] = wmma_f16(a, b, acc[tm]);
            }
        }
        const float bias = proj_b[s * 16 + n];
#pragma unroll
        for (int tm = 0; tm < 4; ++tm)
#pragma unroll
            for (int r = 0; r < 8; ++r)
                outg[(tm * 16 + r + 8 * h) * 256 + s * 16 + n] = acc[tm][r] + bias;
    }
}

extern "C" void kernel_launch(void* const* d_in, const int* in_sizes, int n_in,
                              void* d_out, int out_size, void* d_ws, size_t ws_size,
                              hipStream_t stream) {
    const float* x      = (const float*)d_in[0];
    const float* qkv_w  = (const float*)d_in[1];
    const float* qkv_b  = (const float*)d_in[2];
    const float* proj_w = (const float*)d_in[3];
    const float* proj_b = (const float*)d_in[4];
    float* out = (float*)d_out;

    _Float16* qkvw_h = (_Float16*)d_ws;                 // 768*256 f16 = 384 KB
    _Float16* projw_h = qkvw_h + 768 * 256;             // 256*256 f16 = 128 KB

    convert_weights_kernel<<<256, 256, 0, stream>>>(qkv_w, proj_w, qkvw_h, projw_h);

    const int NW = in_sizes[0] / (64 * 256);            // 4096 windows
    WindowAttention_54168127537767_kernel<<<NW, 256, 131072, stream>>>(
        x, qkvw_h, qkv_b, projw_h, proj_b, out);
}
